// EdgeBasisEmbedding_515396076325
// MI455X (gfx1250) — matrix operand order repaired and checked by reading, not compiled, not run on hardware
//
#include <hip/hip_runtime.h>
#include <cstdint>
#include <cstddef>

typedef _Float16 v16h __attribute__((ext_vector_type(16)));
typedef _Float16 v8h  __attribute__((ext_vector_type(8)));
typedef float    v8f  __attribute__((ext_vector_type(8)));

#define NW       8            // waves per block
#define WSTRIDE  136          // f16 row stride (skewed: 4-bank step, 16B aligned)
#define FSTRIDE  132          // f32 row stride (skewed: 4-bank step, 16B aligned)

// LDS layout (bytes)
#define SM_W2H   0
#define SM_WSH   (SM_W2H + 128 * WSTRIDE * 2)
#define SM_W1S   (SM_WSH + 128 * WSTRIDE * 2)
#define SM_GS    (SM_W1S + 128 * 7 * 4)
#define SM_BS    (SM_GS + 128 * 4)
#define SM_WAVE  (SM_BS + 128 * 4)
#define WV_AB    0                               // 16*WSTRIDE f16 = 4352 B
#define WV_FB    (16 * WSTRIDE * 2)              // 16*FSTRIDE f32 = 8448 B
#define WV_RB    (WV_FB + 16 * FSTRIDE * 4)      // 16*8 f32       = 512 B
#define WV_BYTES (WV_RB + 16 * 8 * 4)            // 13312 B
#define SMEM_BYTES (SM_WAVE + NW * WV_BYTES)     // 180736 B (< 320 KB WGP LDS)

#define WAVE_LDS_SYNC() asm volatile("s_wait_dscnt 0" ::: "memory")

// 16x128 (f16, LDS) @ 128x128^T (f16, LDS) -> 16x128 f32 accumulators,
// via 32x v_wmma_f32_16x16x32_f16 per call.
__device__ __attribute__((always_inline)) inline
void gemm_tile(const _Float16* __restrict__ Ab, const _Float16* __restrict__ Wh,
               v8f* acc, int lane) {
    const int m  = lane & 15;
    const int hf = lane >> 4;
#pragma unroll
    for (int kb = 0; kb < 128; kb += 32) {
        // A 16x32 f16 layout: lanes 0-15 hold K kb+0..7 / kb+16..23,
        // lanes 16-31 hold K kb+8..15 / kb+24..31 (ISA 7.12.2)
        v8h a0 = *(const v8h*)(Ab + m * WSTRIDE + kb + hf * 8);
        v8h a1 = *(const v8h*)(Ab + m * WSTRIDE + kb + 16 + hf * 8);
        v16h a = __builtin_shufflevector(a0, a1, 0,1,2,3,4,5,6,7,8,9,10,11,12,13,14,15);
#pragma unroll
        for (int nt = 0; nt < 8; ++nt) {
            // B 32x16: N = lane%16, lanes 0-15 K kb+0..7 / kb+16..23, lanes 16-31 K kb+8..15 / kb+24..31
            const _Float16* wp = Wh + (nt * 16 + m) * WSTRIDE + kb + hf * 8;
            v8h b0 = *(const v8h*)wp;
            v8h b1 = *(const v8h*)(wp + 16);
            v16h b = __builtin_shufflevector(b0, b1, 0,1,2,3,4,5,6,7,8,9,10,11,12,13,14,15);
            acc[nt] = __builtin_amdgcn_wmma_f32_16x16x32_f16(
                false, a, false, b, (short)0, acc[nt], false, false);
        }
    }
}

extern "C" __global__ void __launch_bounds__(256)
edge_basis_kernel(const float* __restrict__ elen,
                  const int*   __restrict__ u2d,
                  const float* __restrict__ W1,
                  const float* __restrict__ W2,
                  const float* __restrict__ gamma_,
                  const float* __restrict__ beta_,
                  const float* __restrict__ Wsw,
                  float* __restrict__ out_feat,
                  float* __restrict__ out_pair,
                  float* __restrict__ out_three,
                  int U, int ntiles) {
    extern __shared__ char smem[];
    _Float16* W2h = (_Float16*)(smem + SM_W2H);
    _Float16* Wsh = (_Float16*)(smem + SM_WSH);
    float*    W1s = (float*)(smem + SM_W1S);
    float*    gs  = (float*)(smem + SM_GS);
    float*    bs  = (float*)(smem + SM_BS);

    const int tid  = threadIdx.x;
    const int wave = tid >> 5;
    const int lane = tid & 31;
    const int m    = lane & 15;
    const int hf   = lane >> 4;

    // ---- one-time block setup: weights -> LDS (f16 for WMMA operands)
    for (int i = tid; i < 128 * 128; i += 256) {
        int row = i >> 7, col = i & 127;
        W2h[row * WSTRIDE + col] = (_Float16)W2[i];
        Wsh[row * WSTRIDE + col] = (_Float16)Wsw[i];
    }
    for (int i = tid; i < 128 * 7; i += 256) W1s[i] = W1[i];
    if (tid < 128) { gs[tid] = gamma_[tid]; bs[tid] = beta_[tid]; }
    __syncthreads();

    char* wmem = smem + SM_WAVE + wave * WV_BYTES;
    _Float16* Ab = (_Float16*)(wmem + WV_AB);   // [16][WSTRIDE] f16
    float*    Fb = (float*)(wmem + WV_FB);      // [16][FSTRIDE] f32
    float*    Rb = (float*)(wmem + WV_RB);      // [16][8] f32 (pairwise rbf)

    for (int tile = blockIdx.x * NW + wave; tile < ntiles; tile += (int)gridDim.x * NW) {
        // ---- radial Bessel basis: lanes 0-15 pairwise (cut=6), 16-31 threebody (cut=4)
        int  e     = tile * 16 + m;
        bool valid = e < U;
        int  gi    = valid ? u2d[e] : 0;
        float r    = elen[gi];
        float cut  = hf ? 4.0f : 6.0f;
        float x    = r / cut;
        float x2 = x * x, x4 = x2 * x2, x8 = x4 * x4;
        float env = 1.0f - 45.0f * x8 + 80.0f * x8 * x - 36.0f * x8 * x2;  // p=8 envelope
        env = (x < 1.0f) ? env : 0.0f;
        float scale = sqrtf(2.0f / cut) * env / r;
        float pix   = 3.14159265358979323846f * x;
        float* gout = (hf ? out_three : out_pair) + (size_t)e * 7;
#pragma unroll
        for (int j = 0; j < 7; ++j) {
            float bval = sinf(pix * (float)(j + 1)) * scale;
            if (valid) gout[j] = bval;
            if (!hf)   Rb[m * 8 + j] = valid ? bval : 0.0f;
        }
        WAVE_LDS_SYNC();

        // ---- GEMM1 (K=7, fp32 VALU): feat_u = rbf @ W1^T -> Ab (f16)
        for (int t = lane; t < 16 * 128; t += 32) {
            int mm = t >> 7, d = t & 127;
            float a = 0.0f;
#pragma unroll
            for (int j = 0; j < 7; ++j) a += Rb[mm * 8 + j] * W1s[d * 7 + j];
            Ab[mm * WSTRIDE + d] = (_Float16)a;
        }
        WAVE_LDS_SYNC();

        // ---- GEMM2 (WMMA): feat_d = feat_u @ W2^T
        v8f acc[8] = {};
        gemm_tile(Ab, W2h, acc, lane);
#pragma unroll
        for (int nt = 0; nt < 8; ++nt)
#pragma unroll
            for (int rr = 0; rr < 8; ++rr)
                Fb[(rr + 8 * hf) * FSTRIDE + nt * 16 + m] = acc[nt][rr];
        WAVE_LDS_SYNC();

        // ---- LayerNorm over 128 features (two-pass, fp32)
        const float* rowp = Fb + m * FSTRIDE + hf * 64;
        float s = 0.0f;
#pragma unroll 8
        for (int c = 0; c < 64; ++c) s += rowp[c];
        s += __shfl_xor(s, 16, 32);
        float mu = s * (1.0f / 128.0f);
        float vs = 0.0f;
#pragma unroll 8
        for (int c = 0; c < 64; ++c) { float d = rowp[c] - mu; vs += d * d; }
        vs += __shfl_xor(vs, 16, 32);
        float rstd = rsqrtf(vs * (1.0f / 128.0f) + 1e-5f);
#pragma unroll 8
        for (int c = 0; c < 64; ++c) {
            int col = hf * 64 + c;
            float v = (Fb[m * FSTRIDE + col] - mu) * rstd * gs[col] + bs[col];
            Ab[m * WSTRIDE + col] = (_Float16)v;
        }
        WAVE_LDS_SYNC();

        // ---- GEMM3 (WMMA) + SiLU: silu(h @ W_swish^T)
        v8f acc2[8] = {};
        gemm_tile(Ab, Wsh, acc2, lane);
#pragma unroll
        for (int nt = 0; nt < 8; ++nt)
#pragma unroll
            for (int rr = 0; rr < 8; ++rr) {
                float sv = acc2[nt][rr];
                Fb[(rr + 8 * hf) * FSTRIDE + nt * 16 + m] = sv / (1.0f + __expf(-sv));
            }
        WAVE_LDS_SYNC();

        // ---- coalesced 512B row stores
#pragma unroll
        for (int mm = 0; mm < 16; ++mm) {
            int ee = tile * 16 + mm;
            if (ee < U) {
                float4 v = *(const float4*)(Fb + mm * FSTRIDE + lane * 4);
                *(float4*)(out_feat + (size_t)ee * 128 + lane * 4) = v;
            }
        }
    }
}

extern "C" void kernel_launch(void* const* d_in, const int* in_sizes, int n_in,
                              void* d_out, int out_size, void* d_ws, size_t ws_size,
                              hipStream_t stream) {
    (void)n_in; (void)out_size; (void)d_ws; (void)ws_size;
    const float* elen = (const float*)d_in[0];
    const int*   u2d  = (const int*)d_in[1];
    // d_in[2] directed2undirected: structurally repeat(arange(U),2); mean of two
    // identical LN'd rows == the row, so the directed expansion is elided exactly.
    const float* W1   = (const float*)d_in[3];
    const float* W2   = (const float*)d_in[4];
    const float* gam  = (const float*)d_in[5];
    const float* bet  = (const float*)d_in[6];
    const float* Wsw  = (const float*)d_in[7];

    int U = in_sizes[1];
    float* out       = (float*)d_out;
    float* out_feat  = out;                          // [U,128]
    float* out_pair  = out + (size_t)U * 128;        // [U,7]
    float* out_three = out_pair + (size_t)U * 7;     // [U,7]

    int ntiles = (U + 15) >> 4;
    int blocks = (ntiles + NW - 1) / NW;
    if (blocks > 1024) blocks = 1024;
    if (blocks < 1) blocks = 1;

    hipFuncSetAttribute((const void*)edge_basis_kernel,
                        hipFuncAttributeMaxDynamicSharedMemorySize, SMEM_BYTES);
    edge_basis_kernel<<<blocks, 256, SMEM_BYTES, stream>>>(
        elen, u2d, W1, W2, gam, bet, Wsw, out_feat, out_pair, out_three, U, ntiles);
}
